// PilotAEQ_DP_26001732010262
// MI455X (gfx1250) — matrix operand order repaired and checked by reading, not compiled, not run on hardware
//
#include <hip/hip_runtime.h>

// ---------------- problem constants ----------------
#define FFT_N   524288      // 2^19
#define LOG2N   19
#define HALF_N  262144      // N/2
#define NPV     131072
#define NP_UPV  262144
#define LV      31
#define EQ_OFFV 7           // (L-1)/2 / SPS
#define N_OUT   262128      // (N - L) / 2
#define N_ADAPT 65529       // i*2 + 14 < NP  =>  i < 65529  (last updating step)
#define N_P1    65532       // phase-1 end, rounded so phase-2 start is 16B-aligned
#define OUTC    262228      // 100 + N_OUT
#define LRV     0.001f
#define RED_BLOCKS 1024
#define NPAD    (FFT_N + 64)   // padded half-plane length (OOB-safe async tiles)

typedef __attribute__((ext_vector_type(16))) _Float16 v16h;
typedef __attribute__((ext_vector_type(8)))  float    v8f;

// ---------------- helpers ----------------
__device__ __forceinline__ float2 cmulconj(float2 a, float2 b) {
    return make_float2(a.x * b.x + a.y * b.y, a.y * b.x - a.x * b.y);   // a * conj(b)
}

// ---------------- stage 0: pack [y0,y1,p0,p1] into batch-4 FFT input ----------------
__global__ __launch_bounds__(256) void pack_fft_input(const float2* __restrict__ y,
                                                      const float2* __restrict__ pup,
                                                      float2* __restrict__ dst) {
    int tid = blockIdx.x * 256 + threadIdx.x;      // < 4*N
    int b = tid >> LOG2N;
    int n = tid & (FFT_N - 1);
    float2 v = make_float2(0.f, 0.f);
    if (b < 2)            v = y[tid];
    else if (n < NP_UPV)  v = pup[(b - 2) * NP_UPV + n];
    dst[tid] = v;
}

// ---------------- radix-2 Stockham pass (batch 4), sgn=+1 fwd, -1 inv ----------------
__global__ __launch_bounds__(256) void fft_pass(const float2* __restrict__ src,
                                                float2* __restrict__ dst,
                                                int log2Ns, float sgn) {
    int tid = blockIdx.x * 256 + threadIdx.x;      // < 4*HALF_N
    int b = tid >> (LOG2N - 1);
    int t = tid & (HALF_N - 1);
    const float2* s = src + ((size_t)b << LOG2N);
    float2*       d = dst + ((size_t)b << LOG2N);
    int Ns = 1 << log2Ns;
    int k  = t & (Ns - 1);
    float ang = sgn * (-3.14159265358979f) * (float)k / (float)Ns;
    float si, co;
    __sincosf(ang, &si, &co);
    float2 u = s[t];
    float2 v = s[t + HALF_N];
    float2 wv = make_float2(co * v.x - si * v.y, co * v.y + si * v.x);
    int j = ((t >> log2Ns) << (log2Ns + 1)) + k;
    d[j]      = make_float2(u.x + wv.x, u.y + wv.y);
    d[j + Ns] = make_float2(u.x - wv.x, u.y - wv.y);
}

// ---------------- spectrum products: C_ij = Y_i * conj(P_j) ----------------
__global__ __launch_bounds__(256) void cross_mult(const float2* __restrict__ F,
                                                  float2* __restrict__ C) {
    int n = blockIdx.x * 256 + threadIdx.x;        // < N
    float2 y0 = F[n];
    float2 y1 = F[FFT_N + n];
    float2 p0 = F[2 * (size_t)FFT_N + n];
    float2 p1 = F[3 * (size_t)FFT_N + n];
    C[n]                     = cmulconj(y0, p0);
    C[FFT_N + n]             = cmulconj(y0, p1);
    C[2 * (size_t)FFT_N + n] = cmulconj(y1, p0);
    C[3 * (size_t)FFT_N + n] = cmulconj(y1, p1);
}

// ---------------- |.|-combine + partial max/argmax ----------------
__global__ __launch_bounds__(256) void corr_reduce1(const float2* __restrict__ C,
                                                    float2* __restrict__ pmax,
                                                    int2* __restrict__ pidx) {
    float bd = -1.f, bc = -1.f;
    int   id = 0, ic = 0;
    for (int n = blockIdx.x * 256 + threadIdx.x; n < FFT_N; n += gridDim.x * 256) {
        float2 c00 = C[n];
        float2 c01 = C[FFT_N + n];
        float2 c10 = C[2 * (size_t)FFT_N + n];
        float2 c11 = C[3 * (size_t)FFT_N + n];
        float dir = sqrtf(c00.x * c00.x + c00.y * c00.y) + sqrtf(c11.x * c11.x + c11.y * c11.y);
        float crs = sqrtf(c01.x * c01.x + c01.y * c01.y) + sqrtf(c10.x * c10.x + c10.y * c10.y);
        if (dir > bd) { bd = dir; id = n; }
        if (crs > bc) { bc = crs; ic = n; }
    }
    __shared__ float sd[256], sc[256];
    __shared__ int   sid[256], sic[256];
    int t = threadIdx.x;
    sd[t] = bd; sc[t] = bc; sid[t] = id; sic[t] = ic;
    __syncthreads();
    for (int off = 128; off > 0; off >>= 1) {
        if (t < off) {
            if (sd[t + off] > sd[t] || (sd[t + off] == sd[t] && sid[t + off] < sid[t])) {
                sd[t] = sd[t + off]; sid[t] = sid[t + off];
            }
            if (sc[t + off] > sc[t] || (sc[t + off] == sc[t] && sic[t + off] < sic[t])) {
                sc[t] = sc[t + off]; sic[t] = sic[t + off];
            }
        }
        __syncthreads();
    }
    if (t == 0) {
        pmax[blockIdx.x] = make_float2(sd[0], sc[0]);
        pidx[blockIdx.x] = make_int2(sid[0], sic[0]);
    }
}

// ---------------- final reduction -> meta = {start, swap} ----------------
__global__ __launch_bounds__(256) void corr_reduce2(const float2* __restrict__ pmax,
                                                    const int2* __restrict__ pidx,
                                                    int* __restrict__ meta) {
    float bd = -1.f, bc = -1.f;
    int   id = 0, ic = 0;
    for (int n = threadIdx.x; n < RED_BLOCKS; n += 256) {
        float2 v = pmax[n];
        int2   w = pidx[n];
        if (v.x > bd || (v.x == bd && w.x < id)) { bd = v.x; id = w.x; }
        if (v.y > bc || (v.y == bc && w.y < ic)) { bc = v.y; ic = w.y; }
    }
    __shared__ float sd[256], sc[256];
    __shared__ int   sid[256], sic[256];
    int t = threadIdx.x;
    sd[t] = bd; sc[t] = bc; sid[t] = id; sic[t] = ic;
    __syncthreads();
    for (int off = 128; off > 0; off >>= 1) {
        if (t < off) {
            if (sd[t + off] > sd[t] || (sd[t + off] == sd[t] && sid[t + off] < sid[t])) {
                sd[t] = sd[t + off]; sid[t] = sid[t + off];
            }
            if (sc[t + off] > sc[t] || (sc[t + off] == sc[t] && sic[t + off] < sic[t])) {
                sc[t] = sc[t + off]; sic[t] = sic[t + off];
            }
        }
        __syncthreads();
    }
    if (t == 0) {
        int swap = (sc[0] > sd[0]) ? 1 : 0;
        meta[0] = swap ? sic[0] : sid[0];
        meta[1] = swap;
    }
}

// ---------------- roll (f32, for the adaptive phase) ----------------
__global__ __launch_bounds__(256) void roll_k(const float2* __restrict__ y,
                                              const int* __restrict__ meta,
                                              float2* __restrict__ yc) {
    int tid = blockIdx.x * 256 + threadIdx.x;      // < 2*N
    int pol = tid >> LOG2N;
    int n   = tid & (FFT_N - 1);
    int start = meta[0];
    int sw    = meta[1];
    int sp    = sw ? (1 - pol) : pol;
    yc[((size_t)pol << LOG2N) + n] = y[((size_t)sp << LOG2N) + ((n + start) & (FFT_N - 1))];
}

// ---------------- roll (f16, de-interleaved planes for the WMMA phase) ----------------
// ych layout: plane r = pol*2 + (0:re,1:im), each NPAD halves (64-sample zero pad).
__global__ __launch_bounds__(256) void roll_half(const float2* __restrict__ y,
                                                 const int* __restrict__ meta,
                                                 _Float16* __restrict__ ych) {
    int tid = blockIdx.x * 256 + threadIdx.x;
    if (tid >= 2 * NPAD) return;
    int pol = tid / NPAD;
    int n   = tid - pol * NPAD;
    float2 v = make_float2(0.f, 0.f);
    if (n < FFT_N) {
        int start = meta[0];
        int sw    = meta[1];
        int sp    = sw ? (1 - pol) : pol;
        v = y[((size_t)sp << LOG2N) + ((n + start) & (FFT_N - 1))];
    }
    ych[(size_t)(pol * 2 + 0) * NPAD + n] = (_Float16)v.x;
    ych[(size_t)(pol * 2 + 1) * NPAD + n] = (_Float16)v.y;
}

// ---------------- leading 100 zero columns of the output ----------------
__global__ void zero_head(float2* __restrict__ out) {
    int t = threadIdx.x;
    if (t < 200) {
        int pol = t / 100, c = t % 100;
        out[(size_t)pol * OUTC + c] = make_float2(0.f, 0.f);
    }
}

// ---------------- phase 1: serial LMS (one wave32, lane-per-tap) ----------------
__global__ __launch_bounds__(32) void eq_adapt(const float2* __restrict__ yc,
                                               const float2* __restrict__ pilot,
                                               const float2* __restrict__ taps0,
                                               float2* __restrict__ out,
                                               float2* __restrict__ tapsF) {
    int  k   = threadIdx.x;        // tap index (lane 31 idle)
    bool act = (k < LV);
    float t0re = 0.f, t0im = 0.f, t1re = 0.f, t1im = 0.f;
    float t2re = 0.f, t2im = 0.f, t3re = 0.f, t3im = 0.f;
    if (act) {
        float2 a = taps0[0 * LV + k]; t0re = a.x; t0im = a.y;
        float2 b = taps0[1 * LV + k]; t1re = b.x; t1im = b.y;
        float2 c = taps0[2 * LV + k]; t2re = c.x; t2im = c.y;
        float2 d = taps0[3 * LV + k]; t3re = d.x; t3im = d.y;
    }
    const float TWO_LR = 2.0f * LRV;
#pragma unroll 1
    for (int i = 0; i < N_P1; ++i) {
        int pos = i * 2 + 30 - k;  // reversed window
        float2 w0 = make_float2(0.f, 0.f), w1 = make_float2(0.f, 0.f);
        if (act) {
            w0 = yc[pos];
            w1 = yc[FFT_N + pos];
            __builtin_prefetch((const void*)&yc[pos + 16], 0, 3);
            __builtin_prefetch((const void*)&yc[FFT_N + pos + 16], 0, 3);
        }
        float s0re = t0re * w0.x - t0im * w0.y + t1re * w1.x - t1im * w1.y;
        float s0im = t0re * w0.y + t0im * w0.x + t1re * w1.y + t1im * w1.x;
        float s1re = t2re * w1.x - t2im * w1.y + t3re * w0.x - t3im * w0.y;
        float s1im = t2re * w1.y + t2im * w1.x + t3re * w0.y + t3im * w0.x;
        for (int off = 16; off > 0; off >>= 1) {
            s0re += __shfl_xor(s0re, off, 32);
            s0im += __shfl_xor(s0im, off, 32);
            s1re += __shfl_xor(s1re, off, 32);
            s1im += __shfl_xor(s1im, off, 32);
        }
        if (i < N_ADAPT) {
            float2 p0 = pilot[EQ_OFFV + i];
            float2 p1 = pilot[NPV + EQ_OFFV + i];
            float e0re = p0.x - s0re, e0im = p0.y - s0im;
            float e1re = p1.x - s1re, e1im = p1.y - s1im;
            t0re += TWO_LR * (e0re * w0.x + e0im * w0.y);
            t0im += TWO_LR * (e0im * w0.x - e0re * w0.y);
            t1re += TWO_LR * (e0re * w1.x + e0im * w1.y);
            t1im += TWO_LR * (e0im * w1.x - e0re * w1.y);
            t2re += TWO_LR * (e1re * w1.x + e1im * w1.y);
            t2im += TWO_LR * (e1im * w1.x - e1re * w1.y);
            t3re += TWO_LR * (e1re * w0.x + e1im * w0.y);
            t3im += TWO_LR * (e1im * w0.x - e1re * w0.y);
        }
        if (k == 0) {
            out[100 + i]                = make_float2(s0re, s0im);
            out[(size_t)OUTC + 100 + i] = make_float2(s1re, s1im);
        }
    }
    if (act) {
        tapsF[0 * LV + k] = make_float2(t0re, t0im);
        tapsF[1 * LV + k] = make_float2(t1re, t1im);
        tapsF[2 * LV + k] = make_float2(t2re, t2im);
        tapsF[3 * LV + k] = make_float2(t3re, t3im);
    }
}

// ---------------- phase 2 fragment builders (all /31 math folds at compile time) ----------------
// Feature vector (ascending j, reversed taps): x[i][sec*31+j] = plane_sec(y_cut)[2i+j]
// A matrix rows 0..3 = [Re0, Im0, Re1, Im1]; A[m][sec*31+j] built from taps[.., 30-j].
template<int S, int HI>
__device__ __forceinline__ v16h fragA(const float2* __restrict__ tapsF, int m) {
    v16h a;
#pragma unroll
    for (int h = 0; h < 16; ++h) {
        const int kin = (HI ? 16 : 0) + h;
        const int k   = 32 * S + kin;
        float v = 0.f;
        if (k < 124) {
            const int sec = k / 31;            // compile-time
            const int j   = k - sec * 31;      // compile-time
            const int pp  = sec >> 1;          // 0 -> taps{0,3}, 1 -> taps{1,2}
            if (m < 4) {
                int f = (m < 2) ? (pp ? 1 : 0) : (pp ? 2 : 3);
                float2 tp = tapsF[f * LV + (30 - j)];
                if ((sec & 1) == 0) v = (m & 1) ? tp.y : tp.x;
                else                v = (m & 1) ? tp.x : -tp.y;
            }
        }
        a[h] = (_Float16)v;
    }
    return a;
}

template<int KB>
__device__ __forceinline__ v16h fragB(const _Float16* __restrict__ smrow, int n) {
    v16h b;
#pragma unroll
    for (int h = 0; h < 16; ++h) {
        const int k = KB + h;
        _Float16 v = (_Float16)0.0f;
        if (k < 124) {
            const int sec = k / 31;            // compile-time
            const int j   = k - sec * 31;      // compile-time
            v = smrow[sec * 64 + 2 * n + j];   // affine in n only
        }
        b[h] = v;
    }
    return b;
}

// ---------------- phase 2: frozen-tap MIMO FIR as WMMA GEMM ----------------
#define P2_WAVES 8
__global__ __launch_bounds__(256) void eq_wmma(const _Float16* __restrict__ ych,
                                               const float2* __restrict__ tapsF,
                                               float2* __restrict__ out) {
    // per-wave tile: 4 rows (pol,re/im) x 64 halves = 512 B
    __shared__ __align__(16) _Float16 smd[P2_WAVES][4][64];
    int lane = threadIdx.x & 31;
    int wave = threadIdx.x >> 5;
    const int tiles = (N_OUT - N_P1 + 15) / 16;      // 12288
    int tile = blockIdx.x * P2_WAVES + wave;
    if (tile >= tiles) tile = tiles - 1;             // overhang waves redo last tile
    int i0 = N_P1 + tile * 16;                       // 2*i0 multiple of 8 -> 16B aligned bytes
    int m  = lane & 15;
    int hi = lane >> 4;

    // ---- async-DMA the 4x128B window tile into LDS (one b128 per lane) ----
    {
        int r   = lane >> 3;                         // row 0..3
        int cch = lane & 7;                          // 16B chunk 0..7
        const _Float16* gsrc = ych + (size_t)r * NPAD + 2 * (size_t)i0;
        unsigned long long gaddr = (unsigned long long)(uintptr_t)gsrc
                                 + (unsigned long long)(cch * 16);
        unsigned ldsoff = (unsigned)(uintptr_t)&smd[wave][0][0]
                        + (unsigned)(r * 128 + cch * 16);
        asm volatile("global_load_async_to_lds_b128 %0, %1, off"
                     :: "v"(ldsoff), "v"(gaddr) : "memory");
    }

    // ---- A fragments (built from global taps while DMA is in flight) ----
    v16h a0, a1, a2, a3;
    if (hi == 0) {
        a0 = fragA<0, 0>(tapsF, m); a1 = fragA<1, 0>(tapsF, m);
        a2 = fragA<2, 0>(tapsF, m); a3 = fragA<3, 0>(tapsF, m);
    } else {
        a0 = fragA<0, 1>(tapsF, m); a1 = fragA<1, 1>(tapsF, m);
        a2 = fragA<2, 1>(tapsF, m); a3 = fragA<3, 1>(tapsF, m);
    }

    asm volatile("s_wait_asynccnt 0x0" ::: "memory");

    // ---- B fragments + 4 chained WMMAs ----
    const _Float16* smrow = &smd[wave][0][0];
    int n = m;
    v8f c = {};
    if (hi == 0) {
        v16h b;
        b = fragB<  0>(smrow, n);
        c = __builtin_amdgcn_wmma_f32_16x16x32_f16(false, a0, false, b, (short)0, c, false, false);
        b = fragB< 32>(smrow, n);
        c = __builtin_amdgcn_wmma_f32_16x16x32_f16(false, a1, false, b, (short)0, c, false, false);
        b = fragB< 64>(smrow, n);
        c = __builtin_amdgcn_wmma_f32_16x16x32_f16(false, a2, false, b, (short)0, c, false, false);
        b = fragB< 96>(smrow, n);
        c = __builtin_amdgcn_wmma_f32_16x16x32_f16(false, a3, false, b, (short)0, c, false, false);
    } else {
        v16h b;
        b = fragB< 16>(smrow, n);
        c = __builtin_amdgcn_wmma_f32_16x16x32_f16(false, a0, false, b, (short)0, c, false, false);
        b = fragB< 48>(smrow, n);
        c = __builtin_amdgcn_wmma_f32_16x16x32_f16(false, a1, false, b, (short)0, c, false, false);
        b = fragB< 80>(smrow, n);
        c = __builtin_amdgcn_wmma_f32_16x16x32_f16(false, a2, false, b, (short)0, c, false, false);
        b = fragB<112>(smrow, n);
        c = __builtin_amdgcn_wmma_f32_16x16x32_f16(false, a3, false, b, (short)0, c, false, false);
    }

    // ---- D rows 0..3 = [Re0, Im0, Re1, Im1] for column n (lanes 0-15) ----
    if (hi == 0) {
        int i = i0 + n;
        if (i < N_OUT) {
            out[100 + i]                = make_float2(c[0], c[1]);
            out[(size_t)OUTC + 100 + i] = make_float2(c[2], c[3]);
        }
    }
}

// ---------------- launcher ----------------
extern "C" void kernel_launch(void* const* d_in, const int* in_sizes, int n_in,
                              void* d_out, int out_size, void* d_ws, size_t ws_size,
                              hipStream_t stream) {
    (void)in_sizes; (void)n_in; (void)out_size; (void)ws_size;
    const float2* y     = (const float2*)d_in[0];  // (2, N) complex64
    const float2* pilot = (const float2*)d_in[1];  // (2, NP)
    const float2* pup   = (const float2*)d_in[2];  // (2, NP_UP)
    const float2* taps0 = (const float2*)d_in[3];  // (4, 31)
    float2* out = (float2*)d_out;                  // (2, OUTC)

    char* ws = (char*)d_ws;
    float2*   bufA  = (float2*)(ws);                           // 16 MB
    float2*   bufB  = (float2*)(ws + (size_t)16 * 1024 * 1024);// 16 MB
    float2*   pmax  = (float2*)(ws + (size_t)32 * 1024 * 1024);
    int2*     pidx  = (int2*)  (ws + (size_t)32 * 1024 * 1024 + 8 * 1024);
    int*      meta  = (int*)   (ws + (size_t)32 * 1024 * 1024 + 16 * 1024);
    float2*   tapsF = (float2*)(ws + (size_t)32 * 1024 * 1024 + 16 * 1024 + 64);
    _Float16* ych   = (_Float16*)(ws + (size_t)32 * 1024 * 1024 + 32 * 1024); // 4*NPAD halves ~4MB

    // forward FFT: batch 4 [y0,y1,p0,p1]
    pack_fft_input<<<(4 * FFT_N) / 256, 256, 0, stream>>>(y, pup, bufA);
    float2* pa = bufA;
    float2* pb = bufB;
    for (int p = 0; p < LOG2N; ++p) {
        fft_pass<<<(4 * HALF_N) / 256, 256, 0, stream>>>(pa, pb, p, 1.0f);
        float2* t = pa; pa = pb; pb = t;
    }
    // spectra in pa; products -> pb
    cross_mult<<<FFT_N / 256, 256, 0, stream>>>(pa, pb);
    { float2* t = pa; pa = pb; pb = t; }
    // inverse FFT: batch 4 [c00,c01,c10,c11] (unnormalized; argmax is scale-invariant)
    for (int p = 0; p < LOG2N; ++p) {
        fft_pass<<<(4 * HALF_N) / 256, 256, 0, stream>>>(pa, pb, p, -1.0f);
        float2* t = pa; pa = pb; pb = t;
    }
    // correlations in pa; pb free -> f32 y_cut for the adaptive phase
    float2* ycut = pb;
    corr_reduce1<<<RED_BLOCKS, 256, 0, stream>>>(pa, pmax, pidx);
    corr_reduce2<<<1, 256, 0, stream>>>(pmax, pidx, meta);
    roll_k<<<(2 * FFT_N) / 256, 256, 0, stream>>>(y, meta, ycut);
    roll_half<<<(2 * NPAD + 255) / 256, 256, 0, stream>>>(y, meta, ych);

    zero_head<<<1, 256, 0, stream>>>(out);
    eq_adapt<<<1, 32, 0, stream>>>(ycut, pilot, taps0, out, tapsF);

    const int tiles   = (N_OUT - N_P1 + 15) / 16;              // 12288
    const int nblocks = (tiles + P2_WAVES - 1) / P2_WAVES;     // 1536
    eq_wmma<<<nblocks, 256, 0, stream>>>(ych, tapsF, out);
}
